// Qwen3Attention_50525995270649
// MI455X (gfx1250) — compile-verified, matrix operand order checked by reading, not run on hardware
//
#include <hip/hip_runtime.h>
#include <hip/hip_bf16.h>
#include <math.h>
#include <stdint.h>

#define HIDDEN 2048
#define NUM_HEADS 16
#define NUM_KV_HEADS 8
#define HEAD_DIM 128
#define EPS 1e-6f
#define ATT_SCALE 0.08838834764831845f  // 128^-0.5
#define ROPE_THETA 10000.0f

typedef __bf16 bf16;
typedef __attribute__((ext_vector_type(16))) __bf16 v16bf;
typedef __attribute__((ext_vector_type(8)))  __bf16 v8bf;
typedef __attribute__((ext_vector_type(8)))  float  v8f;
typedef __attribute__((ext_vector_type(4)))  unsigned int u32x4;
typedef __attribute__((ext_vector_type(8)))  int i32x8;
typedef __attribute__((ext_vector_type(4)))  int i32x4;

#if defined(__gfx1250__) && __has_builtin(__builtin_amdgcn_tensor_load_to_lds)
#define HAVE_TDM 1
#else
#define HAVE_TDM 0
#endif

// Load a 16-element bf16 WMMA fragment slice for this lane.
// Per CDNA5 ISA (16-bit A 16x32 / B 32x16): lanes 0-15 hold K={0..7,16..23},
// lanes 16-31 hold K={8..15,24..31}. Caller pre-offsets p by (+8 for hi half).
static __device__ inline v16bf load_frag(const bf16* p) {
  v8bf lo = *(const v8bf*)p;
  v8bf hi = *(const v8bf*)(p + 16);
  return __builtin_shufflevector(lo, hi, 0,1,2,3,4,5,6,7,8,9,10,11,12,13,14,15);
}

static __device__ inline v8f wmma_bf16(v16bf a, v16bf b, v8f c) {
  return __builtin_amdgcn_wmma_f32_16x16x32_bf16(false, a, false, b, (short)0, c, false, false);
}

#if HAVE_TDM
// Issue a 2D TDM tile load: tile_d1 rows x tile_d0 cols (bf16) from global
// (row stride stride_elems) into LDS at lds_addr.
// D# packing per CDNA5 ISA; 6-arg builtin form
// (g0, g1, g2, g3, extra, cpol) per this toolchain's signature.
static __device__ inline void tdm_load_2d(unsigned lds_addr, const void* gptr,
                                          unsigned tile_d0, unsigned tile_d1,
                                          unsigned stride_elems) {
  unsigned long long ga = (unsigned long long)(uintptr_t)gptr;
  const unsigned td0 = 0x40000000u, td1 = 0x40000000u;  // huge dims: no OOB clip
  u32x4 g0;
  g0[0] = 1u;                                            // count=1, user mode
  g0[1] = lds_addr;                                      // lds_addr [63:32]
  g0[2] = (unsigned)ga;                                  // global_addr [95:64]
  g0[3] = (unsigned)((ga >> 32) & 0x1FFFFFFu) | (2u << 30);  // addr[56:32], type=2
  i32x8 g1;
  g1[0] = (int)(1u << 16);                               // data_size=1 -> 2 bytes
  g1[1] = (int)((td0 & 0xFFFFu) << 16);                  // atomic_addr=0, tdim0 lo16
  g1[2] = (int)((td0 >> 16) | ((td1 & 0xFFFFu) << 16));  // tdim0 hi16, tdim1 lo16
  g1[3] = (int)((td1 >> 16) | (tile_d0 << 16));          // tdim1 hi16, tile_dim0
  g1[4] = (int)(tile_d1 & 0xFFFFu);                      // tile_dim1, tile_dim2=0
  g1[5] = (int)stride_elems;                             // dim0 stride lo32
  g1[6] = 0;                                             // dim0 stride hi16, dim1 stride lo16
  g1[7] = 0;                                             // dim1 stride hi32
  i32x4 z4 = {};
  i32x8 z8 = {};
  __builtin_amdgcn_tensor_load_to_lds(g0, g1, z4, z4, z8, 0);
}
#endif

// ---------------- converts ----------------
__global__ void cvt_f32_bf16(const float* __restrict__ in, bf16* __restrict__ out, size_t n) {
  size_t i = (size_t)blockIdx.x * blockDim.x + threadIdx.x;
  size_t stride = (size_t)gridDim.x * blockDim.x;
  for (; i < n; i += stride) out[i] = (bf16)in[i];
}

// in[K][N] f32 -> out[N][K] bf16  (so GEMM B operand columns are contiguous)
__global__ void tcvt_f32_bf16(const float* __restrict__ in, bf16* __restrict__ out, int K, int N) {
  size_t n = (size_t)K * N;
  size_t i = (size_t)blockIdx.x * blockDim.x + threadIdx.x;
  size_t stride = (size_t)gridDim.x * blockDim.x;
  for (; i < n; i += stride) {
    int k = (int)(i / N);
    int nn = (int)(i % N);
    out[(size_t)nn * K + k] = (bf16)in[i];
  }
}

// vf [t][8][128] f32 -> vtb [kvh][d][t] bf16 (transposed for PV B fragments)
__global__ void vtr_kernel(const float* __restrict__ vf, bf16* __restrict__ vtb, int T) {
  size_t n = (size_t)T * NUM_KV_HEADS * HEAD_DIM;
  size_t i = (size_t)blockIdx.x * blockDim.x + threadIdx.x;
  size_t stride = (size_t)gridDim.x * blockDim.x;
  for (; i < n; i += stride) {
    int t = (int)(i / (NUM_KV_HEADS * HEAD_DIM));
    int r = (int)(i % (NUM_KV_HEADS * HEAD_DIM)); // h*128+d
    vtb[(size_t)r * T + t] = (bf16)vf[i];
  }
}

// ---------------- bf16 GEMM: C[M,N] (f32) = A[M,K] x BT[N,K] ----------------
// Block = 8 waves computes a 128x64 C tile. K consumed in 64-wide chunks that
// are double-buffered in LDS; wave 0 drives the Tensor Data Mover (TENSORcnt)
// to prefetch the next chunk while all waves run WMMAs out of the current one.
__global__ __launch_bounds__(256) void gemm_bf16_kernel(
    const bf16* __restrict__ A, const bf16* __restrict__ BT,
    float* __restrict__ C, int M, int N, int K) {
  __shared__ bf16 sA[2][128 * 64];  // [buf][row m][k]   32 KB
  __shared__ bf16 sB[2][64 * 64];   // [buf][row n][k]   16 KB
  int lane = threadIdx.x & 31, wave = threadIdx.x >> 5;
  int tilesN = N >> 6;
  int mb = (int)blockIdx.x / tilesN, nb = (int)blockIdx.x % tilesN;
  int m0 = mb << 7, n0 = nb << 6;
  int lm = lane & 15, hf = lane >> 4;
  v8f acc[4] = {};
  const int nChunks = K >> 6;

#if HAVE_TDM
  if (wave == 0) {
    tdm_load_2d((unsigned)(uintptr_t)&sA[0][0], A + (size_t)m0 * K, 64, 128, (unsigned)K);
    tdm_load_2d((unsigned)(uintptr_t)&sB[0][0], BT + (size_t)n0 * K, 64, 64, (unsigned)K);
  }
#else
  {
    // cooperative copy of chunk 0: 256 threads, 16B each per trip
    int tid = threadIdx.x;
    for (int e = tid; e < 128 * 8; e += 256) {  // A: 128 rows x 64 cols / 8 per v8bf
      int row = e >> 3, c8 = (e & 7) * 8;
      *(v8bf*)&sA[0][row * 64 + c8] = *(const v8bf*)(A + (size_t)(m0 + row) * K + c8);
    }
    for (int e = tid; e < 64 * 8; e += 256) {
      int row = e >> 3, c8 = (e & 7) * 8;
      *(v8bf*)&sB[0][row * 64 + c8] = *(const v8bf*)(BT + (size_t)(n0 + row) * K + c8);
    }
  }
#endif

  for (int c = 0; c < nChunks; ++c) {
    int cur = c & 1;
#if HAVE_TDM
    if (wave == 0) {
      if (c + 1 < nChunks) {
        int k1 = (c + 1) << 6;
        tdm_load_2d((unsigned)(uintptr_t)&sA[cur ^ 1][0], A + (size_t)m0 * K + k1, 64, 128, (unsigned)K);
        tdm_load_2d((unsigned)(uintptr_t)&sB[cur ^ 1][0], BT + (size_t)n0 * K + k1, 64, 64, (unsigned)K);
        __builtin_amdgcn_s_wait_tensorcnt(2);  // chunk c complete; next 2 in flight
      } else {
        __builtin_amdgcn_s_wait_tensorcnt(0);
      }
    }
    __syncthreads();  // publish buffer `cur` to all waves
#else
    __syncthreads();
    if (c + 1 < nChunks) {
      int k1 = (c + 1) << 6, tid = threadIdx.x, nxt = cur ^ 1;
      for (int e = tid; e < 128 * 8; e += 256) {
        int row = e >> 3, c8 = (e & 7) * 8;
        *(v8bf*)&sA[nxt][row * 64 + c8] = *(const v8bf*)(A + (size_t)(m0 + row) * K + k1 + c8);
      }
      for (int e = tid; e < 64 * 8; e += 256) {
        int row = e >> 3, c8 = (e & 7) * 8;
        *(v8bf*)&sB[nxt][row * 64 + c8] = *(const v8bf*)(BT + (size_t)(n0 + row) * K + k1 + c8);
      }
    }
#endif
    const bf16* a0 = &sA[cur][(wave * 16 + lm) * 64 + (hf ? 8 : 0)];
    const bf16* b0 = &sB[cur][lm * 64 + (hf ? 8 : 0)];
    #pragma unroll
    for (int ks = 0; ks < 2; ++ks) {
      v16bf a = load_frag(a0 + ks * 32);
      #pragma unroll
      for (int j = 0; j < 4; ++j) {
        v16bf b = load_frag(b0 + j * 16 * 64 + ks * 32);
        acc[j] = wmma_bf16(a, b, acc[j]);
      }
    }
    __syncthreads();  // done reading buffer `cur`; safe to overwrite next trip
  }

  #pragma unroll
  for (int j = 0; j < 4; ++j)
    #pragma unroll
    for (int r = 0; r < 8; ++r)
      C[(size_t)(m0 + wave * 16 + r + 8 * hf) * N + (n0 + j * 16 + lm)] = acc[j][r];
}

// ---------------- fused per-head RMSNorm + RoPE, f32 -> bf16 ----------------
__global__ __launch_bounds__(128) void normrope_kernel(
    const float* __restrict__ src, const float* __restrict__ w,
    const int* __restrict__ positions, bf16* __restrict__ dst, int T, int nh) {
  int lane = threadIdx.x & 31;
  int idx = (int)((blockIdx.x * blockDim.x + threadIdx.x) >> 5);
  if (idx >= T * nh) return;
  int t = idx / nh;
  const float* p = src + (size_t)idx * HEAD_DIM;
  float x0 = p[lane], x1 = p[lane + 32], x2 = p[lane + 64], x3 = p[lane + 96];
  float ss = x0 * x0 + x1 * x1 + x2 * x2 + x3 * x3;
  #pragma unroll
  for (int off = 1; off < 32; off <<= 1) ss += __shfl_xor(ss, off, 32);
  float inv = rsqrtf(ss * (1.0f / HEAD_DIM) + EPS);
  x0 *= inv * w[lane];      x1 *= inv * w[lane + 32];
  x2 *= inv * w[lane + 64]; x3 *= inv * w[lane + 96];
  float pos = (float)positions[t];
  float f0 = pos * powf(ROPE_THETA, -(2.0f * lane) / HEAD_DIM);
  float f1 = pos * powf(ROPE_THETA, -(2.0f * (lane + 32)) / HEAD_DIM);
  float c0 = cosf(f0), s0 = sinf(f0), c1 = cosf(f1), s1 = sinf(f1);
  bf16* q = dst + (size_t)idx * HEAD_DIM;
  q[lane]      = (bf16)(x0 * c0 - x2 * s0);
  q[lane + 64] = (bf16)(x2 * c0 + x0 * s0);
  q[lane + 32] = (bf16)(x1 * c1 - x3 * s1);
  q[lane + 96] = (bf16)(x3 * c1 + x1 * s1);
}

// ---------------- flash attention: one wave per (head, 16-query tile) ----------------
// qb [t][16][128] bf16, kb [t][8][128] bf16, vtb [kvh][128][T] bf16, out f32 [t][16][128]
__global__ __launch_bounds__(128) void flash_attn_kernel(
    const bf16* __restrict__ qb, const bf16* __restrict__ kb,
    const bf16* __restrict__ vtb, float* __restrict__ attn, int T) {
  __shared__ bf16 lds_p[4 * 16 * 32];  // per-wave 16x32 P staging
  int lane = threadIdx.x & 31, wave = threadIdx.x >> 5;
  int gw = blockIdx.x * 4 + wave;
  int h = gw % NUM_HEADS, qt = gw / NUM_HEADS;
  if (qt >= (T >> 4)) return;
  int qbase = qt * 16;
  int lm = lane & 15, hf = lane >> 4;
  int kvh = h >> 1;  // GQA repeat factor 2

  v16bf aq[4];
  const bf16* qrow = qb + ((size_t)(qbase + lm) * NUM_HEADS + h) * HEAD_DIM + (hf ? 8 : 0);
  #pragma unroll
  for (int ks = 0; ks < 4; ++ks) aq[ks] = load_frag(qrow + ks * 32);

  float mrun[8], lrun[8], alpha[8];
  v8f acc[8] = {};
  #pragma unroll
  for (int r = 0; r < 8; ++r) { mrun[r] = -1e30f; lrun[r] = 0.0f; }

  bf16* pl = lds_p + wave * 512;
  int smax = qbase + 15;
  for (int s0 = 0; s0 <= smax; s0 += 32) {
    v8f c0 = {}, c1 = {};
    const bf16* kr0 = kb + ((size_t)(s0 + lm)      * NUM_KV_HEADS + kvh) * HEAD_DIM + (hf ? 8 : 0);
    const bf16* kr1 = kb + ((size_t)(s0 + 16 + lm) * NUM_KV_HEADS + kvh) * HEAD_DIM + (hf ? 8 : 0);
    #pragma unroll
    for (int ks = 0; ks < 4; ++ks) {
      c0 = wmma_bf16(aq[ks], load_frag(kr0 + ks * 32), c0);
      c1 = wmma_bf16(aq[ks], load_frag(kr1 + ks * 32), c1);
    }
    #pragma unroll
    for (int r = 0; r < 8; ++r) {
      int t = qbase + r + 8 * hf;
      float v0 = ((s0 + lm)      <= t) ? c0[r] * ATT_SCALE : -1e30f;
      float v1 = ((s0 + 16 + lm) <= t) ? c1[r] * ATT_SCALE : -1e30f;
      float m = fmaxf(v0, v1);
      #pragma unroll
      for (int off = 1; off < 16; off <<= 1) m = fmaxf(m, __shfl_xor(m, off, 16));
      float mnew = fmaxf(mrun[r], m);
      float a = __expf(mrun[r] - mnew);
      float p0 = __expf(v0 - mnew);
      float p1 = __expf(v1 - mnew);
      float srow = p0 + p1;
      #pragma unroll
      for (int off = 1; off < 16; off <<= 1) srow += __shfl_xor(srow, off, 16);
      lrun[r] = lrun[r] * a + srow;
      mrun[r] = mnew;
      alpha[r] = a;
      int rr = r + 8 * hf;
      pl[rr * 32 + lm]      = (bf16)p0;
      pl[rr * 32 + 16 + lm] = (bf16)p1;
    }
    #pragma unroll
    for (int j = 0; j < 8; ++j)
      #pragma unroll
      for (int r = 0; r < 8; ++r) acc[j][r] *= alpha[r];
    // reload P in A-fragment layout (same-wave DS ops are in-order)
    v16bf pf = load_frag(pl + lm * 32 + (hf ? 8 : 0));
    const bf16* vcol = vtb + ((size_t)(kvh * HEAD_DIM + lm)) * T + s0 + (hf ? 8 : 0);
    #pragma unroll
    for (int j = 0; j < 8; ++j)
      acc[j] = wmma_bf16(pf, load_frag(vcol + (size_t)(j * 16) * T), acc[j]);
  }
  #pragma unroll
  for (int j = 0; j < 8; ++j)
    #pragma unroll
    for (int r = 0; r < 8; ++r) {
      int t = qbase + r + 8 * hf;
      attn[((size_t)t * NUM_HEADS + h) * HEAD_DIM + j * 16 + lm] = acc[j][r] / lrun[r];
    }
}

// ---------------- launch ----------------
extern "C" void kernel_launch(void* const* d_in, const int* in_sizes, int n_in,
                              void* d_out, int out_size, void* d_ws, size_t ws_size,
                              hipStream_t stream) {
  const float* x   = (const float*)d_in[0];
  const int*   pos = (const int*)d_in[1];
  const float* wq  = (const float*)d_in[2];
  const float* wk  = (const float*)d_in[3];
  const float* wv  = (const float*)d_in[4];
  const float* wo  = (const float*)d_in[5];
  const float* qnw = (const float*)d_in[6];
  const float* knw = (const float*)d_in[7];
  float* out = (float*)d_out;
  const int T = in_sizes[0] / HIDDEN;  // 4096

  char* ws = (char*)d_ws;
  size_t off = 0;
  auto take = [&](size_t bytes) { char* p = ws + off; off += (bytes + 255) & ~(size_t)255; return p; };
  bf16*  xb   = (bf16*) take((size_t)T * HIDDEN * 2);
  bf16*  wqT  = (bf16*) take((size_t)HIDDEN * 2048 * 2);
  bf16*  wkT  = (bf16*) take((size_t)HIDDEN * 1024 * 2);
  bf16*  wvT  = (bf16*) take((size_t)HIDDEN * 1024 * 2);
  bf16*  woT  = (bf16*) take((size_t)2048 * HIDDEN * 2);
  float* qf   = (float*)take((size_t)T * 2048 * 4);
  float* kf   = (float*)take((size_t)T * 1024 * 4);
  float* vf   = (float*)take((size_t)T * 1024 * 4);
  bf16*  qb   = (bf16*) take((size_t)T * 2048 * 2);
  bf16*  kb   = (bf16*) take((size_t)T * 1024 * 2);
  bf16*  vtb  = (bf16*) take((size_t)NUM_KV_HEADS * HEAD_DIM * T * 2);
  float* attnf = qf;          // qf dead after normrope(q)
  bf16*  attnb = (bf16*)kf;   // kf dead after normrope(k)

  auto nblk = [](size_t n) { return (unsigned)((n + 255) / 256); };

  // 1) precision conversions
  cvt_f32_bf16 <<<nblk((size_t)T * HIDDEN), 256, 0, stream>>>(x,  xb,  (size_t)T * HIDDEN);
  tcvt_f32_bf16<<<nblk((size_t)HIDDEN * 2048), 256, 0, stream>>>(wq, wqT, HIDDEN, 2048);
  tcvt_f32_bf16<<<nblk((size_t)HIDDEN * 1024), 256, 0, stream>>>(wk, wkT, HIDDEN, 1024);
  tcvt_f32_bf16<<<nblk((size_t)HIDDEN * 1024), 256, 0, stream>>>(wv, wvT, HIDDEN, 1024);
  tcvt_f32_bf16<<<nblk((size_t)2048 * HIDDEN), 256, 0, stream>>>(wo, woT, 2048, HIDDEN);

  // 2) QKV projections (TDM-staged WMMA)
  gemm_bf16_kernel<<<(T / 128) * (2048 / 64), 256, 0, stream>>>(xb, wqT, qf, T, 2048, HIDDEN);
  gemm_bf16_kernel<<<(T / 128) * (1024 / 64), 256, 0, stream>>>(xb, wkT, kf, T, 1024, HIDDEN);
  gemm_bf16_kernel<<<(T / 128) * (1024 / 64), 256, 0, stream>>>(xb, wvT, vf, T, 1024, HIDDEN);

  // 3) RMSNorm + RoPE -> bf16; V transpose -> bf16
  normrope_kernel<<<(T * NUM_HEADS) / 4,    128, 0, stream>>>(qf, qnw, pos, qb, T, NUM_HEADS);
  normrope_kernel<<<(T * NUM_KV_HEADS) / 4, 128, 0, stream>>>(kf, knw, pos, kb, T, NUM_KV_HEADS);
  vtr_kernel<<<nblk((size_t)T * NUM_KV_HEADS * HEAD_DIM), 256, 0, stream>>>(vf, vtb, T);

  // 4) causal flash attention (WMMA)
  flash_attn_kernel<<<(T / 16) * NUM_HEADS / 4, 128, 0, stream>>>(qb, kb, vtb, attnf, T);

  // 5) output projection (TDM-staged WMMA)
  cvt_f32_bf16<<<nblk((size_t)T * 2048), 256, 0, stream>>>(attnf, attnb, (size_t)T * 2048);
  gemm_bf16_kernel<<<(T / 128) * (2048 / 64), 256, 0, stream>>>(attnb, woT, out, T, 2048, 2048);
}
